// SparsTriangularUpdate_18631568130624
// MI455X (gfx1250) — matrix operand order repaired and checked by reading, not compiled, not run on hardware
//
#include <hip/hip_runtime.h>
#include <hip/hip_bf16.h>

typedef __attribute__((ext_vector_type(16))) _Float16 v16h;
typedef __attribute__((ext_vector_type(8)))  _Float16 v8h;
typedef __attribute__((ext_vector_type(4)))  _Float16 v4h;
typedef __attribute__((ext_vector_type(8)))  float    v8f;
typedef int v4i_vs __attribute__((vector_size(16)));   // matches builtin param type

#define NROWS 524288
#define CDIM  128
#define P1_WGS 1024          // phase-1 workgroups
#define P1_ITER 4            // row-block iterations per WG (1024*4*128 = 524288 rows)
#define NSLOTS (P1_WGS * 8)  // per-wave partial-S slots

// async-copy builtins want int4 pointers in explicit address spaces
#define ASG(p) ((__attribute__((address_space(1))) v4i_vs*)(p))
#define ASL(p) ((__attribute__((address_space(3))) v4i_vs*)(p))

__device__ __forceinline__ float sigmoidf_(float x) {
    return 1.0f / (1.0f + __expf(-x));
}

__device__ __forceinline__ float wave_sum(float v) {
#pragma unroll
    for (int off = 16; off >= 1; off >>= 1)
        v += __shfl_xor(v, off, 32);
    return v;
}

__device__ __forceinline__ void lds_wait() {
    // drain LDS ops before cross-lane reads / handing tiles to the async engine
    asm volatile("s_wait_dscnt 0" ::: "memory");
}

__device__ __forceinline__ void async_wait() {
    // drain async LDS<->global DMA before reusing tiles / reading async-loaded data
    asm volatile("s_wait_asynccnt 0" ::: "memory");
}

// ---------------------------------------------------------------------------
// Kernel 0: swizzle six 128x128 f32 weight matrices into f16 WMMA B-fragments.
// half index = ((((mat*4+kk)*8+nt)*32+lane)*16+j)
// element = W[k][n], n = nt*16 + (lane&15), k = kk*32 + (j&7) + 8*(lane>>4) + 16*(j>>3)
// ---------------------------------------------------------------------------
__global__ __launch_bounds__(256) void prep_weights(
    const float* __restrict__ w_a,  const float* __restrict__ w_ga,
    const float* __restrict__ w_b,  const float* __restrict__ w_gb,
    const float* __restrict__ w_go, const float* __restrict__ w_o,
    _Float16* __restrict__ w16)
{
    int tid = blockIdx.x * 256 + threadIdx.x;   // 98304 total
    int j   = tid & 15;
    int l   = (tid >> 4) & 31;
    int nt  = (tid >> 9) & 7;
    int kk  = (tid >> 12) & 3;
    int mat = tid >> 14;
    int n = nt * 16 + (l & 15);
    int k = kk * 32 + (j & 7) + 8 * (l >> 4) + 16 * (j >> 3);
    const float* src = (mat == 0) ? w_a : (mat == 1) ? w_ga : (mat == 2) ? w_b
                     : (mat == 3) ? w_gb : (mat == 4) ? w_go : w_o;
    w16[tid] = (_Float16)src[k * CDIM + n];
}

// ---------------------------------------------------------------------------
// Kernel 1: layernorm -> a/b gated GEMMs (WMMA f16) -> a (f16) + zn16 via
//           async-from-LDS stores, per-wave partial column sums of b.
// ---------------------------------------------------------------------------
__global__ __launch_bounds__(256) void phase1(
    const float* __restrict__ z,
    const float* __restrict__ ln_g, const float* __restrict__ ln_b,
    const float* __restrict__ b_a,  const float* __restrict__ b_ga,
    const float* __restrict__ b_b,  const float* __restrict__ b_gb,
    const _Float16* __restrict__ w16,
    _Float16* __restrict__ abuf, _Float16* __restrict__ zn16,
    float* __restrict__ P2)
{
    __shared__ _Float16 s_zn[8][16][CDIM];   // 32 KB, per-wave zn tiles
    __shared__ _Float16 s_a [8][16][CDIM];   // 32 KB, per-wave a tiles

    const int lane = threadIdx.x & 31;
    const int w    = threadIdx.x >> 5;
    const int l16  = lane & 15;
    const int lhi  = lane >> 4;

    float g4[4], be4[4];
#pragma unroll
    for (int i = 0; i < 4; ++i) { g4[i] = ln_g[lane * 4 + i]; be4[i] = ln_b[lane * 4 + i]; }

    v8f S_acc[8];
#pragma unroll
    for (int i = 0; i < 8; ++i) S_acc[i] = {};

    for (int t = 0; t < P1_ITER; ++t) {
        const long rowbase = ((long)(t * P1_WGS + blockIdx.x)) * 128 + (long)w * 16;

        // previous iteration's async stores read s_zn/s_a: wait before reuse
        async_wait();

        // ---- layernorm 16 rows; lane owns cols [lane*4, lane*4+4) ----
        for (int m = 0; m < 16; ++m) {
            const float* zr = z + (rowbase + m) * CDIM;
            float4 v = *(const float4*)(zr + lane * 4);
            float mean = wave_sum(v.x + v.y + v.z + v.w) * (1.0f / 128.0f);
            float d0 = v.x - mean, d1 = v.y - mean, d2 = v.z - mean, d3 = v.w - mean;
            float var = wave_sum(d0 * d0 + d1 * d1 + d2 * d2 + d3 * d3) * (1.0f / 128.0f);
            float r = rsqrtf(var + 1e-5f);
            v4h h;
            h[0] = (_Float16)(d0 * r * g4[0] + be4[0]);
            h[1] = (_Float16)(d1 * r * g4[1] + be4[1]);
            h[2] = (_Float16)(d2 * r * g4[2] + be4[2]);
            h[3] = (_Float16)(d3 * r * g4[3] + be4[3]);
            *(v4h*)(&s_zn[w][m][lane * 4]) = h;
        }
        lds_wait();

        // ---- zn16 out: coalesced async DMA from the LDS tile ----
#pragma unroll
        for (int i = 0; i < 8; ++i) {
            __builtin_amdgcn_global_store_async_from_lds_b128(
                ASG(zn16 + (rowbase + i * 2 + lhi) * CDIM + l16 * 8),
                ASL(&s_zn[w][i * 2 + lhi][l16 * 8]), 0, 0);
        }

        // ---- A fragments (16x32 f16 per k-chunk) from LDS ----
        v16h A[4];
#pragma unroll
        for (int kk = 0; kk < 4; ++kk) {
            const _Float16* base = &s_zn[w][l16][0];
            int k0 = kk * 32 + lhi * 8;
            v8h lo = *(const v8h*)(base + k0);
            v8h hi = *(const v8h*)(base + k0 + 16);
#pragma unroll
            for (int i = 0; i < 8; ++i) { A[kk][i] = lo[i]; A[kk][8 + i] = hi[i]; }
        }

        // ---- 4 GEMMs: value/gate for a and b, 8 n-tiles ----
        for (int nt = 0; nt < 8; ++nt) {
            v8f va = {}, vga = {}, vb = {}, vgb = {};
#pragma unroll
            for (int kk = 0; kk < 4; ++kk) {
                size_t fb = ((size_t)kk * 8 + nt) * 512 + (size_t)lane * 16;
                v16h Ba  = *(const v16h*)(w16 + 0 * 16384 + fb);
                v16h Bga = *(const v16h*)(w16 + 1 * 16384 + fb);
                v16h Bb  = *(const v16h*)(w16 + 2 * 16384 + fb);
                v16h Bgb = *(const v16h*)(w16 + 3 * 16384 + fb);
                va  = __builtin_amdgcn_wmma_f32_16x16x32_f16(false, A[kk], false, Ba,  (short)0, va,  false, false);
                vga = __builtin_amdgcn_wmma_f32_16x16x32_f16(false, A[kk], false, Bga, (short)0, vga, false, false);
                vb  = __builtin_amdgcn_wmma_f32_16x16x32_f16(false, A[kk], false, Bb,  (short)0, vb,  false, false);
                vgb = __builtin_amdgcn_wmma_f32_16x16x32_f16(false, A[kk], false, Bgb, (short)0, vgb, false, false);
            }
            int n = nt * 16 + l16;
            float ba_ = b_a[n], bga_ = b_ga[n], bb_ = b_b[n], bgb_ = b_gb[n];
#pragma unroll
            for (int r = 0; r < 8; ++r) {
                float av = sigmoidf_(vga[r] + bga_) * (va[r] + ba_);
                s_a[w][r + 8 * lhi][n] = (_Float16)av;           // D-layout scatter -> LDS
                S_acc[nt][r] += sigmoidf_(vgb[r] + bgb_) * (vb[r] + bb_);
            }
        }
        lds_wait();

        // ---- a out: coalesced async DMA from the LDS tile ----
#pragma unroll
        for (int i = 0; i < 8; ++i) {
            __builtin_amdgcn_global_store_async_from_lds_b128(
                ASG(abuf + (rowbase + i * 2 + lhi) * CDIM + l16 * 8),
                ASL(&s_a[w][i * 2 + lhi][l16 * 8]), 0, 0);
        }
    }

    // ---- dump per-wave partial S (rows m=0..15 -> k = 16*(w&1)+m) ----
    size_t slot = (size_t)blockIdx.x * 8 + w;
#pragma unroll
    for (int nt = 0; nt < 8; ++nt)
#pragma unroll
        for (int r = 0; r < 8; ++r)
            P2[slot * 2048 + (size_t)(r + 8 * lhi) * CDIM + nt * 16 + l16] = S_acc[nt][r];
    // s_endpgm has an implicit wait-idle, so outstanding async stores drain.
}

// ---------------------------------------------------------------------------
// Kernel 2: fixed-order reduction of wave partials -> S[32][128]
// ---------------------------------------------------------------------------
__global__ __launch_bounds__(256) void reduce_S(const float* __restrict__ P2,
                                                float* __restrict__ S)
{
    int tid = blockIdx.x * 256 + threadIdx.x;   // 4096
    int k = tid >> 7, c = tid & 127;
    int par = k >> 4, m = k & 15;
    float s = 0.0f;
    for (int slot = par; slot < NSLOTS; slot += 2)
        s += P2[(size_t)slot * 2048 + (size_t)m * CDIM + c];
    S[k * CDIM + c] = s;
}

// ---------------------------------------------------------------------------
// Kernel 3: kv = a * S[row%32]; layernorm; out = sigmoid(zn@w_go+b)*(kvn@w_o+b)
// ---------------------------------------------------------------------------
__global__ __launch_bounds__(256) void phase2(
    const _Float16* __restrict__ abuf, const _Float16* __restrict__ zn16,
    const float* __restrict__ S,
    const float* __restrict__ ln_g, const float* __restrict__ ln_b,
    const float* __restrict__ b_go, const float* __restrict__ b_o,
    const _Float16* __restrict__ w16,
    float* __restrict__ out)
{
    __shared__ _Float16 s_kv[8][16][CDIM];   // 32 KB
    __shared__ _Float16 s_zn[8][16][CDIM];   // 32 KB (async-loaded zn tiles)

    const int lane = threadIdx.x & 31;
    const int w    = threadIdx.x >> 5;
    const int l16  = lane & 15;
    const int lhi  = lane >> 4;

    const long rowbase = ((long)blockIdx.x * 8 + w) * 16;
    const int  kbase   = (int)(rowbase & 16);   // rowbase % 32 in {0,16}

    // ---- kick off coalesced async DMA of the zn tile into LDS ----
#pragma unroll
    for (int i = 0; i < 8; ++i) {
        __builtin_amdgcn_global_load_async_to_lds_b128(
            ASG(const_cast<_Float16*>(zn16) + (rowbase + i * 2 + lhi) * CDIM + l16 * 8),
            ASL(&s_zn[w][i * 2 + lhi][l16 * 8]), 0, 0);
    }

    float g4[4], be4[4];
#pragma unroll
    for (int i = 0; i < 4; ++i) { g4[i] = ln_g[lane * 4 + i]; be4[i] = ln_b[lane * 4 + i]; }

    // ---- kv + layernorm -> f16 LDS tile (overlaps the async zn DMA) ----
    for (int m = 0; m < 16; ++m) {
        v4h ah = *(const v4h*)(abuf + (rowbase + m) * CDIM + lane * 4);
        float4 s4 = *(const float4*)(S + (kbase + m) * CDIM + lane * 4);
        float k0 = (float)ah[0] * s4.x, k1 = (float)ah[1] * s4.y;
        float k2 = (float)ah[2] * s4.z, k3 = (float)ah[3] * s4.w;
        float mean = wave_sum(k0 + k1 + k2 + k3) * (1.0f / 128.0f);
        float d0 = k0 - mean, d1 = k1 - mean, d2 = k2 - mean, d3 = k3 - mean;
        float var = wave_sum(d0 * d0 + d1 * d1 + d2 * d2 + d3 * d3) * (1.0f / 128.0f);
        float r = rsqrtf(var + 1e-5f);
        v4h h;
        h[0] = (_Float16)(d0 * r * g4[0] + be4[0]);
        h[1] = (_Float16)(d1 * r * g4[1] + be4[1]);
        h[2] = (_Float16)(d2 * r * g4[2] + be4[2]);
        h[3] = (_Float16)(d3 * r * g4[3] + be4[3]);
        *(v4h*)(&s_kv[w][m][lane * 4]) = h;
    }
    lds_wait();

    // ---- A fragments: kv from LDS ----
    v16h Akv[4], Azn[4];
#pragma unroll
    for (int kk = 0; kk < 4; ++kk) {
        int k0 = kk * 32 + lhi * 8;
        const _Float16* kb = &s_kv[w][l16][0];
        v8h lo = *(const v8h*)(kb + k0);
        v8h hi = *(const v8h*)(kb + k0 + 16);
#pragma unroll
        for (int i = 0; i < 8; ++i) { Akv[kk][i] = lo[i]; Akv[kk][8 + i] = hi[i]; }
    }

    // ---- A fragments: zn from the async-loaded LDS tile ----
    async_wait();
#pragma unroll
    for (int kk = 0; kk < 4; ++kk) {
        int k0 = kk * 32 + lhi * 8;
        const _Float16* zb = &s_zn[w][l16][0];
        v8h lo = *(const v8h*)(zb + k0);
        v8h hi = *(const v8h*)(zb + k0 + 16);
#pragma unroll
        for (int i = 0; i < 8; ++i) { Azn[kk][i] = lo[i]; Azn[kk][8 + i] = hi[i]; }
    }

    // ---- GEMMs with w_go (mat 4) and w_o (mat 5) ----
    for (int nt = 0; nt < 8; ++nt) {
        v8f vo = {}, vgo = {};
#pragma unroll
        for (int kk = 0; kk < 4; ++kk) {
            size_t fb = ((size_t)kk * 8 + nt) * 512 + (size_t)lane * 16;
            v16h Bgo = *(const v16h*)(w16 + 4 * 16384 + fb);
            v16h Bo  = *(const v16h*)(w16 + 5 * 16384 + fb);
            vgo = __builtin_amdgcn_wmma_f32_16x16x32_f16(false, Azn[kk], false, Bgo, (short)0, vgo, false, false);
            vo  = __builtin_amdgcn_wmma_f32_16x16x32_f16(false, Akv[kk], false, Bo,  (short)0, vo,  false, false);
        }
        int n = nt * 16 + l16;
        float bgo_ = b_go[n], bo_ = b_o[n];
#pragma unroll
        for (int r = 0; r < 8; ++r)
            out[(rowbase + r + 8 * lhi) * CDIM + n] = sigmoidf_(vgo[r] + bgo_) * (vo[r] + bo_);
    }
}

// ---------------------------------------------------------------------------
extern "C" void kernel_launch(void* const* d_in, const int* in_sizes, int n_in,
                              void* d_out, int out_size, void* d_ws, size_t ws_size,
                              hipStream_t stream) {
    const float* z       = (const float*)d_in[0];
    const float* ln_in_g = (const float*)d_in[1];
    const float* ln_in_b = (const float*)d_in[2];
    const float* w_a  = (const float*)d_in[3];
    const float* b_a  = (const float*)d_in[4];
    const float* w_ga = (const float*)d_in[5];
    const float* b_ga = (const float*)d_in[6];
    const float* w_b  = (const float*)d_in[7];
    const float* b_b  = (const float*)d_in[8];
    const float* w_gb = (const float*)d_in[9];
    const float* b_gb = (const float*)d_in[10];
    const float* ln_o_g = (const float*)d_in[11];
    const float* ln_o_b = (const float*)d_in[12];
    const float* w_go = (const float*)d_in[13];
    const float* b_go = (const float*)d_in[14];
    const float* w_o  = (const float*)d_in[15];
    const float* b_o  = (const float*)d_in[16];
    float* out = (float*)d_out;

    // workspace layout
    char* ws = (char*)d_ws;
    _Float16* w16  = (_Float16*)(ws);                       // 192 KB
    float*    S    = (float*)(ws + 0x30000);                // 16 KB
    float*    P2   = (float*)(ws + 0x40000);                // 64 MB
    _Float16* abuf = (_Float16*)(ws + 0x4040000);           // 128 MB (f16)
    _Float16* zn16 = (_Float16*)(ws + 0xC040000);           // 128 MB

    prep_weights<<<384, 256, 0, stream>>>(w_a, w_ga, w_b, w_gb, w_go, w_o, w16);
    phase1<<<P1_WGS, 256, 0, stream>>>(z, ln_in_g, ln_in_b, b_a, b_ga, b_b, b_gb,
                                       w16, abuf, zn16, P2);
    reduce_S<<<16, 256, 0, stream>>>(P2, S);
    phase2<<<NROWS / 128, 256, 0, stream>>>(abuf, zn16, S, ln_o_g, ln_o_b,
                                            b_go, b_o, w16, out);
}
